// CrossViewAttention_14396730376598
// MI455X (gfx1250) — matrix operand order repaired and checked by reading, not compiled, not run on hardware
//
#include <hip/hip_runtime.h>

typedef __bf16 bf16x8  __attribute__((ext_vector_type(8)));
typedef __bf16 bf16x16 __attribute__((ext_vector_type(16)));
typedef float  f32x8   __attribute__((ext_vector_type(8)));

constexpr int kB  = 4;
constexpr int kV  = 6;
constexpr int kC  = 256;
constexpr int kHW = 4096;
constexpr int kNH = 8;
constexpr int kDH = 32;       // kC / kNH
constexpr int kTM = 16;       // pixel tile (WMMA M)
constexpr int kKS = kC / 32;  // 8 K-steps of 32

// Packed-weight geometry: per matrix, 16 n-tiles x 8 k-steps fragments,
// each fragment = 32 lanes x 16 bf16 (1 KB), lane data contiguous (32 B).
constexpr int kFragElems   = 32 * 16;            // 512 bf16 per fragment
constexpr int kMatElems    = 16 * kKS * kFragElems;  // 65536 bf16 per matrix
constexpr int kPackedTotal = 4 * kMatElems;          // 262144 bf16 = 512 KB

// ---- WMMA wrapper: D = A(16x32 bf16) x B(32x16 bf16) + C(16x16 f32) ----
__device__ __forceinline__ f32x8 wmma_bf16(bf16x16 a, bf16x16 b, f32x8 c) {
  return __builtin_amdgcn_wmma_f32_16x16x32_bf16(
      /*neg_a=*/false, a, /*neg_b=*/false, b,
      /*c_mod=*/(short)0, c, /*reuse_a=*/false, /*reuse_b=*/false);
}

// A-fragment from LDS tile laid out [16 pix][256 ch] bf16.
// 16-bit A 16x32: lane L -> M = L%16; elems 0..15 -> K = g*16+0..15
// (g = L>>4), matching the packed B order. Two aligned 16B LDS loads.
__device__ __forceinline__ bf16x16 load_a_lds(const __bf16* base, int ks) {
  const int lane = threadIdx.x & 31;
  const int pix  = lane & 15;
  const int g    = lane >> 4;
  const __bf16* p = base + pix * kC + ks * 32 + g * 16;
  bf16x8 lo = *reinterpret_cast<const bf16x8*>(p);
  bf16x8 hi = *reinterpret_cast<const bf16x8*>(p + 8);
  return __builtin_shufflevector(lo, hi,
      0, 1, 2, 3, 4, 5, 6, 7, 8, 9, 10, 11, 12, 13, 14, 15);
}

// B-fragment from the pre-swizzled bf16 slab: lane's 16 elems are the 32
// contiguous bytes at frag_base + lane*32 -> two global_load_b128, no cvt.
__device__ __forceinline__ bf16x16 load_b_packed(const __bf16* __restrict__ wm,
                                                 int nt, int ks) {
  const int lane = threadIdx.x & 31;
  const __bf16* p = wm + ((nt * kKS + ks) * 32 + lane) * 16;
  bf16x8 lo = *reinterpret_cast<const bf16x8*>(p);
  bf16x8 hi = *reinterpret_cast<const bf16x8*>(p + 8);
  return __builtin_shufflevector(lo, hi,
      0, 1, 2, 3, 4, 5, 6, 7, 8, 9, 10, 11, 12, 13, 14, 15);
}

// C/D 16x16 f32: lane L, reg r -> M = g*8 + r, N = L%16.
__device__ __forceinline__ void store_c_lds_bf16(__bf16* base, int n0, f32x8 c) {
  const int lane = threadIdx.x & 31;
  const int n    = n0 + (lane & 15);
  const int g    = lane >> 4;
#pragma unroll
  for (int r = 0; r < 8; ++r) base[(g * 8 + r) * kC + n] = (__bf16)c[r];
}

__device__ __forceinline__ void store_c_lds_f32(float* base, int n0, f32x8 c) {
  const int lane = threadIdx.x & 31;
  const int n    = n0 + (lane & 15);
  const int g    = lane >> 4;
#pragma unroll
  for (int r = 0; r < 8; ++r) base[(g * 8 + r) * kC + n] = c[r];
}

__device__ __forceinline__ void store_c_global(float* __restrict__ yb, int n0,
                                               f32x8 c) {
  const int lane = threadIdx.x & 31;
  const int n    = n0 + (lane & 15);
  const int g    = lane >> 4;
#pragma unroll
  for (int r = 0; r < 8; ++r) yb[(size_t)n * kHW + g * 8 + r] = c[r];
}

// ---- One-shot pack: fp32 weights -> bf16 B-fragment layout in d_ws ----
// frag element (lane, e) = W[nt*16 + lane%16][ks*32 + (lane>>4)*16 + e]
__global__ __launch_bounds__(256)
void pack_weights(const float* __restrict__ Wq, const float* __restrict__ Wk,
                  const float* __restrict__ Wv, const float* __restrict__ Wo,
                  __bf16* __restrict__ ws) {
  const int idx = blockIdx.x * 256 + threadIdx.x;
  if (idx >= kPackedTotal) return;
  const float* Ws[4] = {Wq, Wk, Wv, Wo};
  const int m      = idx >> 16;           // matrix
  const int rem    = idx & (kMatElems - 1);
  const int frag   = rem >> 9;            // nt*8 + ks
  const int within = rem & (kFragElems - 1);
  const int lane   = within >> 4;
  const int e      = within & 15;
  const int nt     = frag >> 3;
  const int ks     = frag & 7;
  const int n      = nt * 16 + (lane & 15);
  const int k      = ks * 32 + (lane >> 4) * 16 + e;
  ws[idx] = (__bf16)Ws[m][n * kC + k];
}

__global__ __launch_bounds__(256, 1)
void cross_view_attn_fused(const float* __restrict__ x,
                           const __bf16* __restrict__ wpack,
                           float* __restrict__ y) {
  // ---- LDS staging (168 KB of the 320 KB WGP budget) ----
  __shared__ __align__(16) __bf16 xs[kV][kTM][kC];   // 48 KB: x tile, bf16
  __shared__ __align__(16) __bf16 ksm[kV][kTM][kC];  // 48 KB: K proj
  __shared__ __align__(16) __bf16 vsm[kV][kTM][kC];  // 48 KB: V proj
  __shared__ __align__(16) float  qsm[kTM][kC];      // 16 KB: sum_v Q
  __shared__ __align__(16) __bf16 osm[kTM][kC];      //  8 KB: attn out

  const int wg      = blockIdx.x;
  const int b       = wg / (kHW / kTM);
  const int pixbase = (wg % (kHW / kTM)) * kTM;
  const int t       = threadIdx.x;
  const int wave    = t >> 5;
  const int nt0     = wave * 2;    // this wave's two 16-channel n-tiles
  const int n0      = wave * 32;

  const __bf16* Wqp = wpack + 0 * kMatElems;
  const __bf16* Wkp = wpack + 1 * kMatElems;
  const __bf16* Wvp = wpack + 2 * kMatElems;
  const __bf16* Wop = wpack + 3 * kMatElems;

  // Warm L2 for this wave's packed-weight slices (global_prefetch_b8).
  __builtin_prefetch(Wqp + (nt0 * kKS * 32 + (t & 31)) * 16, 0, 1);
  __builtin_prefetch(Wkp + (nt0 * kKS * 32 + (t & 31)) * 16, 0, 1);
  __builtin_prefetch(Wvp + (nt0 * kKS * 32 + (t & 31)) * 16, 0, 1);
  __builtin_prefetch(Wop + (nt0 * kKS * 32 + (t & 31)) * 16, 0, 1);

  // ---- Phase 0: x[b, v, :, pixtile] -> LDS as bf16, [v][pix][ch] ----
  const float* xb = x + (size_t)b * kV * kC * kHW;
  for (int i = t; i < kV * kC * kTM; i += 256) {
    const int p = i & (kTM - 1);
    const int c = (i >> 4) & (kC - 1);
    const int v = i >> 12;  // i / (kC*kTM)
    xs[v][p][c] = (__bf16)xb[((size_t)v * kC + c) * kHW + pixbase + p];
  }
  __syncthreads();

  // ---- Phase 1: Qsum = sum_v (x_v * Wq^T); Wq fragments hoisted so each
  //      packed slice streams from L2 once per workgroup. ----
  {
    f32x8 q0 = {}, q1 = {};
    for (int ks = 0; ks < kKS; ++ks) {
      bf16x16 b0 = load_b_packed(Wqp, nt0, ks);
      bf16x16 b1 = load_b_packed(Wqp, nt0 + 1, ks);
#pragma unroll
      for (int v = 0; v < kV; ++v) {
        bf16x16 a = load_a_lds(&xs[v][0][0], ks);
        q0 = wmma_bf16(a, b0, q0);
        q1 = wmma_bf16(a, b1, q1);
      }
    }
    store_c_lds_f32(&qsm[0][0], n0, q0);
    store_c_lds_f32(&qsm[0][0], n0 + 16, q1);
  }

  // ---- Phase 2: K_v and V_v for all 6 views; 12 persistent accumulators
  //      per wave keep weight traffic at one pass per workgroup. ----
  for (int m = 0; m < 2; ++m) {
    const __bf16* W = (m == 0) ? Wkp : Wvp;
    f32x8 acc[kV][2];
#pragma unroll
    for (int v = 0; v < kV; ++v) { acc[v][0] = {}; acc[v][1] = {}; }
    for (int ks = 0; ks < kKS; ++ks) {
      bf16x16 b0 = load_b_packed(W, nt0, ks);
      bf16x16 b1 = load_b_packed(W, nt0 + 1, ks);
#pragma unroll
      for (int v = 0; v < kV; ++v) {
        bf16x16 a = load_a_lds(&xs[v][0][0], ks);
        acc[v][0] = wmma_bf16(a, b0, acc[v][0]);
        acc[v][1] = wmma_bf16(a, b1, acc[v][1]);
      }
    }
    __bf16* dst = (m == 0) ? &ksm[0][0][0] : &vsm[0][0][0];
#pragma unroll
    for (int v = 0; v < kV; ++v) {
      store_c_lds_bf16(dst + (size_t)v * kTM * kC, n0, acc[v][0]);
      store_c_lds_bf16(dst + (size_t)v * kTM * kC, n0 + 16, acc[v][1]);
    }
  }
  __syncthreads();

  // ---- Phase 3: per-(pixel, head) attention over V=6 views (VALU) ----
  if (t < kTM * kNH) {
    const int pix = t >> 3;
    const int h   = t & (kNH - 1);
    const int cb  = h * kDH;
    const float wq = (1.0f / kV) * 0.17677669529663687f;  // mean * dh^-0.5
    float s[kV];
#pragma unroll
    for (int v = 0; v < kV; ++v) {
      float d = 0.0f;
      for (int j = 0; j < kDH; ++j)
        d += qsm[pix][cb + j] * (float)ksm[v][pix][cb + j];
      s[v] = d * wq;
    }
    float mx = s[0];
#pragma unroll
    for (int v = 1; v < kV; ++v) mx = fmaxf(mx, s[v]);
    float e[kV], sum = 0.0f;
#pragma unroll
    for (int v = 0; v < kV; ++v) { e[v] = __expf(s[v] - mx); sum += e[v]; }
    const float rs = 1.0f / sum;
    for (int j = 0; j < kDH; ++j) {
      float o = 0.0f;
#pragma unroll
      for (int v = 0; v < kV; ++v) o += e[v] * (float)vsm[v][pix][cb + j];
      osm[pix][cb + j] = (__bf16)(o * rs);
    }
  }
  __syncthreads();

  // ---- Phase 4: y = attn_out * Wo^T, stored straight to HBM ----
  {
    f32x8 o0 = {}, o1 = {};
    for (int ks = 0; ks < kKS; ++ks) {
      bf16x16 a  = load_a_lds(&osm[0][0], ks);
      bf16x16 b0 = load_b_packed(Wop, nt0, ks);
      bf16x16 b1 = load_b_packed(Wop, nt0 + 1, ks);
      o0 = wmma_bf16(a, b0, o0);
      o1 = wmma_bf16(a, b1, o1);
    }
    float* ybase = y + (size_t)b * kC * kHW + pixbase;
    store_c_global(ybase, n0, o0);
    store_c_global(ybase, n0 + 16, o1);
  }
}

extern "C" void kernel_launch(void* const* d_in, const int* in_sizes, int n_in,
                              void* d_out, int out_size, void* d_ws,
                              size_t ws_size, hipStream_t stream) {
  const float* x  = (const float*)d_in[0];
  const float* Wq = (const float*)d_in[1];
  const float* Wk = (const float*)d_in[2];
  const float* Wv = (const float*)d_in[3];
  const float* Wo = (const float*)d_in[4];
  float* y = (float*)d_out;
  __bf16* wpack = (__bf16*)d_ws;  // 512 KB packed bf16 weights

  // One-shot deterministic weight pack (re-run every call; stream-ordered).
  const int pack_blocks = (kPackedTotal + 255) / 256;  // 1024
  pack_weights<<<dim3(pack_blocks), dim3(256), 0, stream>>>(Wq, Wk, Wv, Wo,
                                                            wpack);

  const int num_tiles = kB * (kHW / kTM);  // 1024 workgroups
  cross_view_attn_fused<<<dim3(num_tiles), dim3(256), 0, stream>>>(x, wpack, y);
}
